// MergeBlock_9586367005334
// MI455X (gfx1250) — matrix operand order repaired and checked by reading, not compiled, not run on hardware
//
#include <hip/hip_runtime.h>
#include <hip/hip_bf16.h>
#include <math.h>

// ---------------------------------------------------------------------------
// MergeBlock (B=2, N=4160, C=512, HID=2048, H=4 heads, d=128, Hseq=4096)
// bf16 WMMA (v_wmma_f32_16x16x32_bf16) for all GEMMs, fp32 accumulation.
// ---------------------------------------------------------------------------

typedef __attribute__((ext_vector_type(16))) __bf16 v16bf;
typedef __attribute__((ext_vector_type(8)))  float  v8f;

union BF16x16 { v16bf v; uint4 q[2]; };

#define WMMA_BF16(A_, B_, C_) \
  __builtin_amdgcn_wmma_f32_16x16x32_bf16(false, (A_), false, (B_), (short)0, (C_), false, false)

static __device__ __forceinline__ float gelu_f(float x) {
  return 0.5f * x * (1.0f + erff(x * 0.70710678118654752f));
}

// A operand, 16x32 bf16 tile. p points at &A[row*lda + k0] for this lane's row.
// ISA layout: lanes 0-15 hold K {0..7, 16..23}; lanes 16-31 hold K {8..15, 24..31}.
static __device__ __forceinline__ v16bf load_a_ptr(const __bf16* p, int hal) {
  BF16x16 u;
  u.q[0] = *reinterpret_cast<const uint4*>(p + hal * 8);
  u.q[1] = *reinterpret_cast<const uint4*>(p + 16 + hal * 8);
  return u.v;
}

// B operand, 32x16 bf16 tile (reduction-major per lane). p points at
// &W[col*ldb + k0] for this lane's column. lanes 0-15: K {0..7, 16..23}; 16-31: K {8..15, 24..31}.
static __device__ __forceinline__ v16bf load_b_ptr(const __bf16* p, int hal) {
  BF16x16 u;
  u.q[0] = *reinterpret_cast<const uint4*>(p + hal * 8);
  u.q[1] = *reinterpret_cast<const uint4*>(p + 16 + hal * 8);
  return u.v;
}

// row-index mapping: 0 = identity, 1 = seq tokens (b*4160 + n, n<4096),
// 2 = sem tokens (b*4160 + 4096 + s, s<64)
static __device__ __forceinline__ int map_row(int t, int mode) {
  if (mode == 1) return ((t >> 12) * 4160) + (t & 4095);
  if (mode == 2) return ((t >> 6) * 4160) + 4096 + (t & 63);
  return t;
}

// ---------------------------------------------------------------------------
// fp32 -> bf16 conversion
// ---------------------------------------------------------------------------
__global__ __launch_bounds__(256) void f32_to_bf16_kernel(const float* __restrict__ s,
                                                          __bf16* __restrict__ d, int n) {
  int i = blockIdx.x * 256 + threadIdx.x;
  if (i < n) d[i] = (__bf16)s[i];
}

// ---------------------------------------------------------------------------
// LayerNorm over C=512, one block (128 threads) per row, bf16 output
// ---------------------------------------------------------------------------
__global__ __launch_bounds__(128) void ln_bf16_kernel(const float* __restrict__ x,
                                                      const float* __restrict__ w,
                                                      const float* __restrict__ b,
                                                      __bf16* __restrict__ out) {
  const int row = blockIdx.x;
  const int tid = threadIdx.x;
  const float4 v = reinterpret_cast<const float4*>(x + (size_t)row * 512)[tid];
  __shared__ float red[128];

  red[tid] = v.x + v.y + v.z + v.w;
  __syncthreads();
  for (int off = 64; off; off >>= 1) { if (tid < off) red[tid] += red[tid + off]; __syncthreads(); }
  const float mu = red[0] * (1.0f / 512.0f);
  __syncthreads();

  float dx0 = v.x - mu, dx1 = v.y - mu, dx2 = v.z - mu, dx3 = v.w - mu;
  red[tid] = dx0 * dx0 + dx1 * dx1 + dx2 * dx2 + dx3 * dx3;
  __syncthreads();
  for (int off = 64; off; off >>= 1) { if (tid < off) red[tid] += red[tid + off]; __syncthreads(); }
  const float rs = rsqrtf(red[0] * (1.0f / 512.0f) + 1e-5f);

  const int i = tid * 4;
  __bf16* o = out + (size_t)row * 512 + i;
  o[0] = (__bf16)(dx0 * rs * w[i + 0] + b[i + 0]);
  o[1] = (__bf16)(dx1 * rs * w[i + 1] + b[i + 1]);
  o[2] = (__bf16)(dx2 * rs * w[i + 2] + b[i + 2]);
  o[3] = (__bf16)(dx3 * rs * w[i + 3] + b[i + 3]);
}

// ---------------------------------------------------------------------------
// Generic bf16 WMMA GEMM: out[t,f] = sum_k A[map(t),k] * W[f,k] + bias[f]
// Block = 256 threads = 8 waves in 4x2; block tile 128(M) x 64(N);
// each wave: 2x2 grid of 16x16 accumulators, K stepped by 32.
// epi: 0 = bf16 store, 1 = gelu->bf16, 2 = outf = resid + gamma*(acc+bias),
//      3 = KV scatter (f<512 -> K matrix, f>=512 -> V transposed [b,h,d,N])
// ---------------------------------------------------------------------------
static __device__ __forceinline__ void gemm_epi(v8f acc, int mt, int nt, int lane,
                                                int Nout, int epi,
                                                const float* __restrict__ bias,
                                                __bf16* __restrict__ out0,
                                                __bf16* __restrict__ out1,
                                                float* __restrict__ outf,
                                                const float* __restrict__ resid,
                                                const float* __restrict__ gamma,
                                                int out_map) {
  const int hal = lane >> 4;
  const int f = nt + (lane & 15);
  const float bf = bias[f];
#pragma unroll
  for (int r = 0; r < 8; ++r) {
    const int t = mt + r + 8 * hal;
    const float v = acc[r] + bf;
    if (epi == 0) {
      out0[(size_t)t * Nout + f] = (__bf16)v;
    } else if (epi == 1) {
      out0[(size_t)t * Nout + f] = (__bf16)gelu_f(v);
    } else if (epi == 2) {
      const int g = map_row(t, out_map);
      const size_t idx = (size_t)g * 512 + f;
      outf[idx] = resid[idx] + gamma[f] * v;
    } else { // KV scatter
      if (f < 512) {
        out0[(size_t)t * 512 + f] = (__bf16)v;
      } else {
        const int f2 = f - 512;
        const int hh = f2 >> 7, d = f2 & 127;
        const int bb = t / 4160, n = t - bb * 4160;
        out1[((size_t)(bb * 4 + hh) * 128 + d) * 4160 + n] = (__bf16)v;
      }
    }
  }
}

__global__ __launch_bounds__(256) void gemm_bf16_kernel(
    const __bf16* __restrict__ A, int lda, int a_map,
    const __bf16* __restrict__ W, int K,
    const float* __restrict__ bias, int Nout, int epi,
    __bf16* __restrict__ out0, __bf16* __restrict__ out1,
    float* __restrict__ outf, const float* __restrict__ resid,
    const float* __restrict__ gamma, int out_map) {
  const int lane = threadIdx.x & 31;
  const int wv = threadIdx.x >> 5;
  const int hal = lane >> 4;
  const int wm = wv >> 1, wn = wv & 1;
  const int m_base = blockIdx.y * 128 + wm * 32;
  const int n_base = blockIdx.x * 64 + wn * 32;

  const int t0 = m_base + (lane & 15);
  const __bf16* a0p = A + (size_t)map_row(t0, a_map) * lda;
  const __bf16* a1p = A + (size_t)map_row(t0 + 16, a_map) * lda;
  const __bf16* b0p = W + (size_t)(n_base + (lane & 15)) * K;
  const __bf16* b1p = W + (size_t)(n_base + 16 + (lane & 15)) * K;

  v8f c00 = {}, c01 = {}, c10 = {}, c11 = {};
  for (int k0 = 0; k0 < K; k0 += 32) {
    const v16bf a0 = load_a_ptr(a0p + k0, hal);
    const v16bf a1 = load_a_ptr(a1p + k0, hal);
    const v16bf b0 = load_b_ptr(b0p + k0, hal);
    const v16bf b1 = load_b_ptr(b1p + k0, hal);
    c00 = WMMA_BF16(a0, b0, c00);
    c01 = WMMA_BF16(a0, b1, c01);
    c10 = WMMA_BF16(a1, b0, c10);
    c11 = WMMA_BF16(a1, b1, c11);
  }
  gemm_epi(c00, m_base +  0, n_base +  0, lane, Nout, epi, bias, out0, out1, outf, resid, gamma, out_map);
  gemm_epi(c01, m_base +  0, n_base + 16, lane, Nout, epi, bias, out0, out1, outf, resid, gamma, out_map);
  gemm_epi(c10, m_base + 16, n_base +  0, lane, Nout, epi, bias, out0, out1, outf, resid, gamma, out_map);
  gemm_epi(c11, m_base + 16, n_base + 16, lane, Nout, epi, bias, out0, out1, outf, resid, gamma, out_map);
}

// ---------------------------------------------------------------------------
// Flash attention: one wave per 16-query tile; 32-key tiles streamed.
// Q,K: bf16 [b*N, 512] (head-major within row); V: bf16 transposed [b,h,d,N].
// Output: bf16 [b*N, 512].
// ---------------------------------------------------------------------------
__global__ __launch_bounds__(256) void attn_kernel(const __bf16* __restrict__ Q,
                                                   const __bf16* __restrict__ Km,
                                                   const __bf16* __restrict__ Vt,
                                                   __bf16* __restrict__ O) {
  __shared__ __align__(16) __bf16 pbuf[8][16 * 32];
  const int lane = threadIdx.x & 31;
  const int wv = threadIdx.x >> 5;
  const int hal = lane >> 4;
  const int l15 = lane & 15;

  const int qt = blockIdx.x * 8 + wv;          // 260 blocks * 8 waves = 2080 tiles
  const int tile = qt % 260;
  const int bh = qt / 260;
  const int hh = bh & 3;
  const int b = bh >> 2;
  const int n0 = tile * 16;
  const float scale = 0.08838834764831845f;    // 1/sqrt(128)

  const __bf16* Qb = Q + (size_t)(b * 4160 + n0) * 512 + hh * 128;
  const __bf16* Kb = Km + (size_t)(b * 4160) * 512 + hh * 128;
  const __bf16* Vb = Vt + (size_t)(b * 4 + hh) * 128 * 4160;

  v16bf qa[4];
#pragma unroll
  for (int c = 0; c < 4; ++c) qa[c] = load_a_ptr(Qb + (size_t)l15 * 512 + c * 32, hal);

  v8f vzero = {};
  v8f oacc[8];
#pragma unroll
  for (int j = 0; j < 8; ++j) oacc[j] = vzero;
  float mrow[8], lrow[8];
#pragma unroll
  for (int r = 0; r < 8; ++r) { mrow[r] = -3.0e38f; lrow[r] = 0.0f; }

  for (int kb = 0; kb < 4160; kb += 32) {
    // S = Q * K^T for 16 queries x 32 keys
    v8f s0 = vzero, s1 = vzero;
    const __bf16* k0p = Kb + (size_t)(kb + l15) * 512;
    const __bf16* k1p = Kb + (size_t)(kb + 16 + l15) * 512;
#pragma unroll
    for (int c = 0; c < 4; ++c) {
      const v16bf kb0 = load_b_ptr(k0p + c * 32, hal);
      const v16bf kb1 = load_b_ptr(k1p + c * 32, hal);
      s0 = WMMA_BF16(qa[c], kb0, s0);
      s1 = WMMA_BF16(qa[c], kb1, s1);
    }

    // online softmax (rows live per-VGPR across each 16-lane half)
    float p0[8], p1[8], alpha[8];
#pragma unroll
    for (int r = 0; r < 8; ++r) {
      const float a = s0[r] * scale, c2 = s1[r] * scale;
      float vm = fmaxf(a, c2);
#pragma unroll
      for (int off = 8; off; off >>= 1) vm = fmaxf(vm, __shfl_xor(vm, off, 32));
      const float mn = fmaxf(mrow[r], vm);
      alpha[r] = __expf(mrow[r] - mn);
      mrow[r] = mn;
      p0[r] = __expf(a - mn);
      p1[r] = __expf(c2 - mn);
      float rs = p0[r] + p1[r];
#pragma unroll
      for (int off = 8; off; off >>= 1) rs += __shfl_xor(rs, off, 32);
      lrow[r] = lrow[r] * alpha[r] + rs;
    }
#pragma unroll
    for (int j = 0; j < 8; ++j)
#pragma unroll
      for (int r = 0; r < 8; ++r) oacc[j][r] *= alpha[r];

    // transpose P (D layout -> A layout) through per-wave LDS staging
    __bf16* pb = &pbuf[wv][0];
#pragma unroll
    for (int r = 0; r < 8; ++r) {
      const int row = r + 8 * hal;
      pb[row * 32 + l15] = (__bf16)p0[r];
      pb[row * 32 + 16 + l15] = (__bf16)p1[r];
    }
    __builtin_amdgcn_wave_barrier();
    asm volatile("s_wait_dscnt 0x0" ::: "memory");
    const v16bf pa = load_a_ptr(pb + l15 * 32, hal);
    __builtin_amdgcn_wave_barrier();

    // O += P * V  (V pre-transposed: key index contiguous per d row)
#pragma unroll
    for (int j = 0; j < 8; ++j) {
      const v16bf bv = load_b_ptr(Vb + (size_t)(j * 16 + l15) * 4160 + kb, hal);
      oacc[j] = WMMA_BF16(pa, bv, oacc[j]);
    }
  }

#pragma unroll
  for (int j = 0; j < 8; ++j)
#pragma unroll
    for (int r = 0; r < 8; ++r) {
      const int g = b * 4160 + n0 + r + 8 * hal;
      const int f = hh * 128 + j * 16 + l15;
      O[(size_t)g * 512 + f] = (__bf16)(oacc[j][r] / lrow[r]);
    }
}

// ---------------------------------------------------------------------------
// Depthwise conv (k=3, pad=1) along token dim within each 4096-token batch,
// + bias + exact gelu. in/out: bf16 [8192, 2048].
// ---------------------------------------------------------------------------
__global__ __launch_bounds__(256) void dwconv_gelu_kernel(const __bf16* __restrict__ in,
                                                          const float* __restrict__ w,
                                                          const float* __restrict__ bias,
                                                          __bf16* __restrict__ out) {
  const int idx = blockIdx.x * 256 + threadIdx.x;      // 8192*2048 total
  const int c = idx & 2047;
  const int t = idx >> 11;
  const int n = t & 4095;
  const float xc = (float)in[idx];
  const float xl = (n > 0)    ? (float)in[idx - 2048] : 0.0f;
  const float xr = (n < 4095) ? (float)in[idx + 2048] : 0.0f;
  const float y = w[c * 3] * xl + w[c * 3 + 1] * xc + w[c * 3 + 2] * xr + bias[c];
  out[idx] = (__bf16)gelu_f(y);
}

// ---------------------------------------------------------------------------
// Host launch sequence
// ---------------------------------------------------------------------------
extern "C" void kernel_launch(void* const* d_in, const int* in_sizes, int n_in,
                              void* d_out, int out_size, void* d_ws, size_t ws_size,
                              hipStream_t stream) {
  const float* x      = (const float*)d_in[0];
  const float* ln1_w  = (const float*)d_in[1];
  const float* ln1_b  = (const float*)d_in[2];
  const float* ln2_w  = (const float*)d_in[3];
  const float* ln2_b  = (const float*)d_in[4];
  const float* q_w    = (const float*)d_in[5];
  const float* q_b    = (const float*)d_in[6];
  const float* kv_w   = (const float*)d_in[7];
  const float* kv_b   = (const float*)d_in[8];
  const float* proj_w = (const float*)d_in[9];
  const float* proj_b = (const float*)d_in[10];
  const float* fc1_w  = (const float*)d_in[11];
  const float* fc1_b  = (const float*)d_in[12];
  const float* dw_w   = (const float*)d_in[13];
  const float* dw_b   = (const float*)d_in[14];
  const float* fc2_w  = (const float*)d_in[15];
  const float* fc2_b  = (const float*)d_in[16];
  const float* px1_w  = (const float*)d_in[17];
  const float* px1_b  = (const float*)d_in[18];
  const float* px2_w  = (const float*)d_in[19];
  const float* px2_b  = (const float*)d_in[20];
  const float* gamma1 = (const float*)d_in[21];
  const float* gamma2 = (const float*)d_in[22];
  float* out = (float*)d_out;

  const int M = 8320;            // B*N tokens
  char* cur = (char*)d_ws;
  auto alloc = [&](size_t bytes) -> char* {
    char* p = cur;
    cur += (bytes + 255) & ~(size_t)255;
    return p;
  };

  __bf16* wq    = (__bf16*)alloc((size_t)512 * 512 * 2);
  __bf16* wkv   = (__bf16*)alloc((size_t)1024 * 512 * 2);
  __bf16* wproj = (__bf16*)alloc((size_t)512 * 512 * 2);
  __bf16* wfc1  = (__bf16*)alloc((size_t)2048 * 512 * 2);
  __bf16* wfc2  = (__bf16*)alloc((size_t)512 * 2048 * 2);
  __bf16* wpx1  = (__bf16*)alloc((size_t)1024 * 512 * 2);
  __bf16* wpx2  = (__bf16*)alloc((size_t)512 * 1024 * 2);
  __bf16* ln1o  = (__bf16*)alloc((size_t)M * 512 * 2);
  __bf16* qbf   = (__bf16*)alloc((size_t)M * 512 * 2);
  __bf16* kbf   = (__bf16*)alloc((size_t)M * 512 * 2);
  __bf16* vT    = (__bf16*)alloc((size_t)M * 512 * 2);   // [b,h,d,N]
  __bf16* attnb = (__bf16*)alloc((size_t)M * 512 * 2);
  float*  x2    = (float*)alloc((size_t)M * 512 * 4);
  __bf16* ln2o  = (__bf16*)alloc((size_t)M * 512 * 2);
  __bf16* fc1o  = (__bf16*)alloc((size_t)8192 * 2048 * 2);
  __bf16* dwo   = (__bf16*)alloc((size_t)8192 * 2048 * 2);
  __bf16* px1o  = (__bf16*)alloc((size_t)128 * 1024 * 2);

  auto conv = [&](const float* s, __bf16* d, int n) {
    f32_to_bf16_kernel<<<(n + 255) / 256, 256, 0, stream>>>(s, d, n);
  };
  conv(q_w,   wq,    512 * 512);
  conv(kv_w,  wkv,   1024 * 512);
  conv(proj_w, wproj, 512 * 512);
  conv(fc1_w, wfc1,  2048 * 512);
  conv(fc2_w, wfc2,  512 * 2048);
  conv(px1_w, wpx1,  1024 * 512);
  conv(px2_w, wpx2,  512 * 1024);

  // --- attention branch ---
  ln_bf16_kernel<<<M, 128, 0, stream>>>(x, ln1_w, ln1_b, ln1o);

  gemm_bf16_kernel<<<dim3(512 / 64, M / 128), 256, 0, stream>>>(
      ln1o, 512, 0, wq, 512, q_b, 512, /*epi*/0, qbf, nullptr, nullptr, nullptr, nullptr, 0);

  gemm_bf16_kernel<<<dim3(1024 / 64, M / 128), 256, 0, stream>>>(
      ln1o, 512, 0, wkv, 512, kv_b, 1024, /*epi*/3, kbf, vT, nullptr, nullptr, nullptr, 0);

  attn_kernel<<<260, 256, 0, stream>>>(qbf, kbf, vT, attnb);

  // proj + gamma1 residual -> x2 (fp32)
  gemm_bf16_kernel<<<dim3(512 / 64, M / 128), 256, 0, stream>>>(
      attnb, 512, 0, wproj, 512, proj_b, 512, /*epi*/2, nullptr, nullptr, x2, x, gamma1, 0);

  // --- FFN branch ---
  ln_bf16_kernel<<<M, 128, 0, stream>>>(x2, ln2_w, ln2_b, ln2o);

  // seq path: fc1 -> dwconv+gelu -> fc2 (+gamma2 residual to out)
  gemm_bf16_kernel<<<dim3(2048 / 64, 8192 / 128), 256, 0, stream>>>(
      ln2o, 512, /*a_map seq*/1, wfc1, 512, fc1_b, 2048, /*epi*/0, fc1o, nullptr, nullptr, nullptr, nullptr, 0);

  dwconv_gelu_kernel<<<(8192 * 2048) / 256, 256, 0, stream>>>(fc1o, dw_w, dw_b, dwo);

  gemm_bf16_kernel<<<dim3(512 / 64, 8192 / 128), 256, 0, stream>>>(
      dwo, 2048, 0, wfc2, 2048, fc2_b, 512, /*epi*/2, nullptr, nullptr, out, x2, gamma2, /*out_map seq*/1);

  // sem path: px1+gelu -> px2 (+gamma2 residual to out)
  gemm_bf16_kernel<<<dim3(1024 / 64, 1), 256, 0, stream>>>(
      ln2o, 512, /*a_map sem*/2, wpx1, 512, px1_b, 1024, /*epi*/1, px1o, nullptr, nullptr, nullptr, nullptr, 0);

  gemm_bf16_kernel<<<dim3(512 / 64, 1), 256, 0, stream>>>(
      px1o, 1024, 0, wpx2, 1024, px2_b, 512, /*epi*/2, nullptr, nullptr, out, x2, gamma2, /*out_map sem*/2);
}
